// PDEEncoder_48558900249390
// MI455X (gfx1250) — matrix operand order repaired and checked by reading, not compiled, not run on hardware
//
#include <hip/hip_runtime.h>

#define BB   4
#define NN   2048
#define DD   256
#define HH   8
#define DHD  32
#define LL   4
#define BH   32      // BB*HH
#define MTOK 8192    // BB*NN

#define AS_STRIDE 40   // padded LDS row stride (halves) -> conflict-free frags
#define BS_STRIDE 40

typedef __attribute__((ext_vector_type(16))) _Float16 v16h;
typedef __attribute__((ext_vector_type(8)))  float    v8f;

union HFrag { v16h v; uint4 u[2]; };

__device__ __forceinline__ v8f wmma16(v16h a, v16h b, v8f c) {
  // D = A(16x32 f16) * B(32x16 f16) + C(16x16 f32)
  return __builtin_amdgcn_wmma_f32_16x16x32_f16(false, a, false, b, (short)0, c, false, false);
}

// ---------------------------------------------------------------------------
// Weight conversion f32 -> f16
// ---------------------------------------------------------------------------
__global__ void f32_to_f16_kernel(const float* __restrict__ s, _Float16* __restrict__ d, int n) {
  const int i = blockIdx.x * blockDim.x + threadIdx.x;
  if (i < n) d[i] = (_Float16)s[i];
}

// ---------------------------------------------------------------------------
// Input projection: h(B,N,1) @ proj_w(1,D) + proj_b  (outer product)
// ---------------------------------------------------------------------------
__global__ void proj_kernel(const float* __restrict__ h0, const float* __restrict__ pw,
                            const float* __restrict__ pb, float* __restrict__ h32,
                            _Float16* __restrict__ h16) {
  const int i = blockIdx.x * blockDim.x + threadIdx.x;   // MTOK*DD threads
  const int row = i >> 8, d = i & 255;
  const float v = h0[row] * pw[d] + pb[d];
  h32[i] = v;
  h16[i] = (_Float16)v;
}

// ---------------------------------------------------------------------------
// RoPE trig tables (layer-invariant): per token 8 freqs x {cos,sin} x {t,x}
// ---------------------------------------------------------------------------
__global__ void trig_kernel(const float* __restrict__ pos, float* __restrict__ trig) {
  const int i = blockIdx.x * blockDim.x + threadIdx.x;   // MTOK*8 threads
  const int row = i >> 3, j = i & 7;
  const float invf = __expf(-(float)j * 1.1512925464970229f);  // ln(10000)/8
  const float ft = pos[row * 2 + 0] * 64.f * invf;
  const float fx = pos[row * 2 + 1] * 64.f * invf;
  float* t = trig + (size_t)row * 32;
  t[j]      = cosf(ft);
  t[j + 8]  = sinf(ft);
  t[j + 16] = cosf(fx);
  t[j + 24] = sinf(fx);
}

// ---------------------------------------------------------------------------
// Apply 2D RoPE in-place to q and k (f16, head-major [BH][N][32])
// ---------------------------------------------------------------------------
__global__ void rope_kernel(_Float16* __restrict__ qkv, const float* __restrict__ trig) {
  const int i = blockIdx.x * blockDim.x + threadIdx.x;   // BH*NN threads
  const int bh = i >> 11, n = i & (NN - 1);
  const int b = bh >> 3;
  const float* t = trig + ((size_t)b * NN + n) * 32;
  float ct[8], st[8], cx[8], sx[8];
#pragma unroll
  for (int j = 0; j < 8; ++j) { ct[j]=t[j]; st[j]=t[j+8]; cx[j]=t[j+16]; sx[j]=t[j+24]; }
  union Row { uint4 u[4]; _Float16 h[32]; };
  for (int which = 0; which < 2; ++which) {
    _Float16* p = qkv + (size_t)which * BH * NN * DHD + ((size_t)bh * NN + n) * DHD;
    Row r, w;
    r.u[0] = ((const uint4*)p)[0]; r.u[1] = ((const uint4*)p)[1];
    r.u[2] = ((const uint4*)p)[2]; r.u[3] = ((const uint4*)p)[3];
#pragma unroll
    for (int j = 0; j < 8; ++j) {
      const float a0 = (float)r.h[j],      a1 = (float)r.h[j + 8];
      w.h[j]      = (_Float16)(a0 * ct[j] - a1 * st[j]);
      w.h[j + 8]  = (_Float16)(a1 * ct[j] + a0 * st[j]);
      const float b0 = (float)r.h[j + 16], b1 = (float)r.h[j + 24];
      w.h[j + 16] = (_Float16)(b0 * cx[j] - b1 * sx[j]);
      w.h[j + 24] = (_Float16)(b1 * cx[j] + b0 * sx[j]);
    }
    ((uint4*)p)[0] = w.u[0]; ((uint4*)p)[1] = w.u[1];
    ((uint4*)p)[2] = w.u[2]; ((uint4*)p)[3] = w.u[3];
  }
}

// ---------------------------------------------------------------------------
// Generic WMMA GEMM: C = A(MxK f16) * B(KxN f16) [+bias], 64x64 block tile,
// 8 waves. A tile staged via GLOBAL_LOAD_ASYNC_TO_LDS_B128 (ASYNCcnt path),
// B tile staged transposed so fragments are contiguous b128 LDS loads.
// MODE: 0=f32 out, 1=f16 out, 2=f16 relu, 3=qkv scatter
// ---------------------------------------------------------------------------
__device__ __forceinline__ void qkv_scatter(_Float16* qkv, int row, int c, float x) {
  const int b = row >> 11, n = row & (NN - 1);
  const int which = c >> 8, rem = c & 255, head = rem >> 5, dh = rem & 31;
  const int bh = b * HH + head;
  size_t idx;
  if (which == 2)   // v stored transposed: [BH][32][N]
    idx = (size_t)2 * ((size_t)BH * NN * DHD) + ((size_t)bh * DHD + dh) * NN + n;
  else              // q / k: [BH][N][32]
    idx = (size_t)which * ((size_t)BH * NN * DHD) + ((size_t)bh * NN + n) * DHD + dh;
  qkv[idx] = (_Float16)x;
}

template <int MODE>
__global__ __launch_bounds__(256) void gemm_kernel(
    const _Float16* __restrict__ A, const _Float16* __restrict__ Bw,
    const float* __restrict__ bias, void* __restrict__ out,
    int M, int K, int Ncols) {
  (void)M;
  __shared__ _Float16 As[64 * AS_STRIDE];    // [m][k], padded
  __shared__ _Float16 Bst[64 * BS_STRIDE];   // transposed: [n][k], padded
  const int tid  = threadIdx.x;
  const int lane = tid & 31;
  const int wave = tid >> 5;
  const int col  = lane & 15;
  const int hi   = lane >> 4;
  const int m0 = blockIdx.y * 64;
  const int n0 = blockIdx.x * 64;
  const int msub  = (wave & 3) * 16;
  const int nsub0 = (wave >> 2) * 16;
  const int nsub1 = nsub0 + 32;

  v8f acc0 = {}, acc1 = {};
  const int ar = tid >> 2, ac = (tid & 3) * 8;   // A stage: row, col-chunk
  const int br = tid >> 3, bc = (tid & 7) * 8;   // B stage: k-row, n-chunk

  for (int k0 = 0; k0 < K; k0 += 32) {
    // --- A tile 64x32: async copy global -> LDS, 16B per lane ---
    {
      const _Float16* gp = &A[(size_t)(m0 + ar) * K + k0 + ac];
      const unsigned loff = (unsigned)(uintptr_t)&As[ar * AS_STRIDE + ac];
      asm volatile("global_load_async_to_lds_b128 %0, %1, off"
                   :: "v"(loff), "v"(gp) : "memory");
    }
    // --- B tile 32x64: coalesced global load, transpose-scatter into LDS ---
    {
      union { uint4 u; _Float16 h[8]; } gb;
      gb.u = *(const uint4*)&Bw[(size_t)(k0 + br) * Ncols + n0 + bc];
#pragma unroll
      for (int j = 0; j < 8; ++j) Bst[(bc + j) * BS_STRIDE + br] = gb.h[j];
    }
    if (k0 + 32 < K) {   // prefetch next k-step tiles into cache
      __builtin_prefetch(&A[(size_t)(m0 + ar) * K + k0 + 32 + ac], 0, 0);
      __builtin_prefetch(&Bw[(size_t)(k0 + 32 + br) * Ncols + n0 + bc], 0, 0);
    }
    asm volatile("s_wait_asynccnt 0x0" ::: "memory");
    __syncthreads();

    // A fragment (16x32): lane row m, halves h -> K = h + 16*hi (contiguous)
    HFrag a;
    const int am = msub + col, kb = hi * 16;
    a.u[0] = *(const uint4*)&As[am * AS_STRIDE + kb];
    a.u[1] = *(const uint4*)&As[am * AS_STRIDE + 8 + kb];
    // B fragments (32x16): lane col n, halves h -> K = h + 16*hi (contiguous)
    HFrag b0, b1;
    const int nb0 = nsub0 + col, nb1 = nsub1 + col, kf = hi * 16;
    b0.u[0] = *(const uint4*)&Bst[nb0 * BS_STRIDE + kf];
    b0.u[1] = *(const uint4*)&Bst[nb0 * BS_STRIDE + kf + 8];
    b1.u[0] = *(const uint4*)&Bst[nb1 * BS_STRIDE + kf];
    b1.u[1] = *(const uint4*)&Bst[nb1 * BS_STRIDE + kf + 8];
    acc0 = wmma16(a.v, b0.v, acc0);
    acc1 = wmma16(a.v, b1.v, acc1);
    __syncthreads();
  }

  float bv0 = 0.f, bv1 = 0.f;
  if (MODE != 3 && bias != nullptr) {
    bv0 = bias[n0 + nsub0 + col];
    bv1 = bias[n0 + nsub1 + col];
  }
#pragma unroll
  for (int r = 0; r < 8; ++r) {        // C layout: reg r -> row r + 8*hi, lane -> col
    const int row = m0 + msub + r + hi * 8;
    const int c0 = n0 + nsub0 + col, c1 = n0 + nsub1 + col;
    float x0 = acc0[r] + bv0, x1 = acc1[r] + bv1;
    if constexpr (MODE == 0) {
      float* O = (float*)out;
      O[(size_t)row * Ncols + c0] = x0;
      O[(size_t)row * Ncols + c1] = x1;
    } else if constexpr (MODE == 1 || MODE == 2) {
      if constexpr (MODE == 2) { x0 = fmaxf(x0, 0.f); x1 = fmaxf(x1, 0.f); }
      _Float16* O = (_Float16*)out;
      O[(size_t)row * Ncols + c0] = (_Float16)x0;
      O[(size_t)row * Ncols + c1] = (_Float16)x1;
    } else {
      qkv_scatter((_Float16*)out, row, c0, x0);
      qkv_scatter((_Float16*)out, row, c1, x1);
    }
  }
}

// ---------------------------------------------------------------------------
// Flash attention: one wave per (b,h, 16-query tile). Computes S^T = K*Q^T so
// the softmax'd P tiles are already in the A-operand layout for P*V.
// ---------------------------------------------------------------------------
__global__ __launch_bounds__(256) void attn_kernel(const _Float16* __restrict__ qkv,
                                                   _Float16* __restrict__ o16) {
  const int wid  = blockIdx.x * 8 + (threadIdx.x >> 5);
  const int lane = threadIdx.x & 31;
  const int col  = lane & 15;
  const int hi   = lane >> 4;
  const int bh = wid >> 7;            // N/16 = 128 q-tiles per (b,h)
  const int q0 = (wid & 127) << 4;

  const _Float16* qp = qkv;
  const _Float16* kp = qkv + (size_t)BH * NN * DHD;
  const _Float16* vp = qkv + (size_t)2 * BH * NN * DHD;

  // B operand = Q^T (32x16): lane col = query, halves 0-7 -> d = 8*hi+h, 8-15 -> d = 16+8*hi+(h-8)
  HFrag bq;
  {
    const _Float16* p = qp + ((size_t)bh * NN + q0 + col) * DHD + hi * 8;
    bq.u[0] = *(const uint4*)p;
    bq.u[1] = *(const uint4*)(p + 16);
  }

  const float scale = 0.1767766952966369f;   // 1/sqrt(32)
  float mrow = -3.0e38f, lrow = 0.f;
  v8f acc0 = {}, acc1 = {};                  // O accum, dh 0-15 / 16-31

  for (int j0 = 0; j0 < NN; j0 += 32) {
    if (j0 + 32 < NN) {   // prefetch next K/V blocks
      __builtin_prefetch(kp + ((size_t)bh * NN + j0 + 32 + col) * DHD, 0, 0);
      __builtin_prefetch(vp + ((size_t)bh * DHD + col) * NN + j0 + 32, 0, 0);
    }
    // A operands = K tiles (16 keys x 32 dh)
    HFrag ak0, ak1;
    {
      const _Float16* p = kp + ((size_t)bh * NN + j0 + col) * DHD + hi * 8;
      ak0.u[0] = *(const uint4*)p;
      ak0.u[1] = *(const uint4*)(p + 16);
      p += 16 * DHD;
      ak1.u[0] = *(const uint4*)p;
      ak1.u[1] = *(const uint4*)(p + 16);
    }
    v8f z = {};
    v8f st0 = wmma16(ak0.v, bq.v, z);  // S^T keys j0..j0+15: lane=query, reg=key r+8*hi
    v8f st1 = wmma16(ak1.v, bq.v, z);  // keys j0+16..j0+31

    float rawmax = st0[0];
#pragma unroll
    for (int r = 0; r < 8; ++r) { rawmax = fmaxf(rawmax, st0[r]); rawmax = fmaxf(rawmax, st1[r]); }
    float mloc = rawmax * scale;
    mloc = fmaxf(mloc, __shfl_xor(mloc, 16));            // combine lane halves
    const float mnew = fmaxf(mrow, mloc);
    const float corr = __expf(mrow - mnew);

    // P in A layout directly: halves 0-7 <- st0 regs, halves 8-15 <- st1 regs
    HFrag ap;
    float rsum = 0.f;
#pragma unroll
    for (int r = 0; r < 8; ++r) {
      const float p0 = __expf(st0[r] * scale - mnew);
      const float p1 = __expf(st1[r] * scale - mnew);
      rsum += p0 + p1;
      ap.v[r]     = (_Float16)p0;
      ap.v[r + 8] = (_Float16)p1;
    }
    rsum += __shfl_xor(rsum, 16);
    lrow = lrow * corr + rsum;
    mrow = mnew;

#pragma unroll
    for (int r = 0; r < 8; ++r) {       // rescale O accum rows q = r + 8*hi
      const float cr = __shfl(corr, r + hi * 8);
      acc0[r] *= cr;
      acc1[r] *= cr;
    }

    // B operands = V blocks from vT [BH][32][N]: lane col = dh, halves -> key
    HFrag bv0, bv1;
    {
      const _Float16* p = vp + ((size_t)bh * DHD + col) * NN + j0 + hi * 8;
      bv0.u[0] = *(const uint4*)p;
      bv0.u[1] = *(const uint4*)(p + 16);
      p += (size_t)16 * NN;
      bv1.u[0] = *(const uint4*)p;
      bv1.u[1] = *(const uint4*)(p + 16);
    }
    acc0 = wmma16(ap.v, bv0.v, acc0);
    acc1 = wmma16(ap.v, bv1.v, acc1);
  }

  const int b = bh >> 3, hh = bh & 7;
#pragma unroll
  for (int r = 0; r < 8; ++r) {
    const float lr  = __shfl(lrow, r + hi * 8);
    const float inv = 1.f / lr;
    const int n = q0 + r + hi * 8;
    const size_t base = ((size_t)(b * NN + n)) * DD + hh * DHD;
    o16[base + col]      = (_Float16)(acc0[r] * inv);
    o16[base + 16 + col] = (_Float16)(acc1[r] * inv);
  }
}

// ---------------------------------------------------------------------------
// h = LayerNorm(h + add) * g + b ; also refresh f16 copy. One block per row.
// ---------------------------------------------------------------------------
__global__ __launch_bounds__(256) void resid_ln_kernel(
    const float* __restrict__ hin, const float* __restrict__ add,
    const float* __restrict__ g, const float* __restrict__ be,
    float* __restrict__ hout, _Float16* __restrict__ h16) {
  const int row = blockIdx.x, t = threadIdx.x;
  const size_t idx = (size_t)row * DD + t;
  const float x = hin[idx] + add[idx];
  float s = x, sq = x * x;
#pragma unroll
  for (int o = 16; o > 0; o >>= 1) { s += __shfl_xor(s, o); sq += __shfl_xor(sq, o); }
  __shared__ float rs[8], rq[8];
  const int w = t >> 5, ln = t & 31;
  if (ln == 0) { rs[w] = s; rq[w] = sq; }
  __syncthreads();
  if (t < 32) {
    float a  = (t < 8) ? rs[t] : 0.f;
    float b2 = (t < 8) ? rq[t] : 0.f;
#pragma unroll
    for (int o = 4; o > 0; o >>= 1) { a += __shfl_xor(a, o); b2 += __shfl_xor(b2, o); }
    if (t == 0) { rs[0] = a; rq[0] = b2; }
  }
  __syncthreads();
  const float mean = rs[0] * (1.f / DD);
  const float var  = rq[0] * (1.f / DD) - mean * mean;
  const float y = (x - mean) * rsqrtf(var + 1e-5f) * g[t] + be[t];
  hout[idx] = y;
  h16[idx]  = (_Float16)y;
}

// ---------------------------------------------------------------------------
extern "C" void kernel_launch(void* const* d_in, const int* in_sizes, int n_in,
                              void* d_out, int out_size, void* d_ws, size_t ws_size,
                              hipStream_t stream) {
  (void)in_sizes; (void)n_in; (void)out_size; (void)ws_size;
  const float* in_h    = (const float*)d_in[0];
  const float* in_pos  = (const float*)d_in[1];
  const float* proj_w  = (const float*)d_in[2];
  const float* proj_b  = (const float*)d_in[3];
  const float* qkv_w   = (const float*)d_in[4];
  const float* out_w   = (const float*)d_in[5];
  const float* out_b   = (const float*)d_in[6];
  const float* O_w     = (const float*)d_in[7];
  const float* O_b     = (const float*)d_in[8];
  const float* ffn1_w  = (const float*)d_in[9];
  const float* ffn1_b  = (const float*)d_in[10];
  const float* ffn2_w  = (const float*)d_in[11];
  const float* ffn2_b  = (const float*)d_in[12];
  const float* ln1_g   = (const float*)d_in[13];
  const float* ln1_b   = (const float*)d_in[14];
  const float* ln2_g   = (const float*)d_in[15];
  const float* ln2_b   = (const float*)d_in[16];
  const float* final_w = (const float*)d_in[17];

  char* wsp = (char*)d_ws;
  size_t off = 0;
  auto carve = [&](size_t bytes) {
    void* p = wsp + off;
    off += (bytes + 255) & ~(size_t)255;
    return p;
  };
  float*    h32  = (float*)carve((size_t)MTOK * DD * 4);
  _Float16* h16  = (_Float16*)carve((size_t)MTOK * DD * 2);
  _Float16* qkvh = (_Float16*)carve((size_t)3 * BH * NN * DHD * 2);  // q,k [BH][N][32]; vT [BH][32][N]
  _Float16* o16  = (_Float16*)carve((size_t)MTOK * DD * 2);
  _Float16* t16  = (_Float16*)carve((size_t)MTOK * 512 * 2);
  float*    t32  = (float*)carve((size_t)MTOK * DD * 4);
  float*    trig = (float*)carve((size_t)MTOK * 32 * 4);
  _Float16* wq   = (_Float16*)carve((size_t)LL * DD * 3 * DD * 2);
  _Float16* wo   = (_Float16*)carve((size_t)LL * DD * DD * 2);
  _Float16* wOp  = (_Float16*)carve((size_t)LL * DD * DD * 2);
  _Float16* wf1  = (_Float16*)carve((size_t)LL * DD * 2 * DD * 2);
  _Float16* wf2  = (_Float16*)carve((size_t)LL * 2 * DD * DD * 2);
  _Float16* wfin = (_Float16*)carve((size_t)DD * DD * 2);

  auto conv = [&](const float* s, _Float16* d, int n) {
    f32_to_f16_kernel<<<(n + 255) / 256, 256, 0, stream>>>(s, d, n);
  };
  conv(qkv_w,   wq,   LL * DD * 3 * DD);
  conv(out_w,   wo,   LL * DD * DD);
  conv(O_w,     wOp,  LL * DD * DD);
  conv(ffn1_w,  wf1,  LL * DD * 2 * DD);
  conv(ffn2_w,  wf2,  LL * 2 * DD * DD);
  conv(final_w, wfin, DD * DD);

  proj_kernel<<<MTOK, 256, 0, stream>>>(in_h, proj_w, proj_b, h32, h16);
  trig_kernel<<<(MTOK * 8) / 256, 256, 0, stream>>>(in_pos, trig);

  for (int l = 0; l < LL; ++l) {
    gemm_kernel<3><<<dim3(12, 128), 256, 0, stream>>>(
        h16, wq + (size_t)l * DD * 3 * DD, nullptr, qkvh, MTOK, DD, 3 * DD);
    rope_kernel<<<(BH * NN) / 256, 256, 0, stream>>>(qkvh, trig);
    attn_kernel<<<512, 256, 0, stream>>>(qkvh, o16);
    gemm_kernel<1><<<dim3(4, 128), 256, 0, stream>>>(
        o16, wo + (size_t)l * DD * DD, out_b + l * DD, t16, MTOK, DD, DD);
    gemm_kernel<0><<<dim3(4, 128), 256, 0, stream>>>(
        t16, wOp + (size_t)l * DD * DD, O_b + l * DD, t32, MTOK, DD, DD);
    resid_ln_kernel<<<MTOK, 256, 0, stream>>>(h32, t32, ln1_g + l * DD, ln1_b + l * DD, h32, h16);
    gemm_kernel<2><<<dim3(8, 128), 256, 0, stream>>>(
        h16, wf1 + (size_t)l * DD * 2 * DD, ffn1_b + l * 2 * DD, t16, MTOK, DD, 2 * DD);
    gemm_kernel<0><<<dim3(4, 128), 256, 0, stream>>>(
        t16, wf2 + (size_t)l * 2 * DD * DD, ffn2_b + l * DD, t32, MTOK, 2 * DD, DD);
    resid_ln_kernel<<<MTOK, 256, 0, stream>>>(h32, t32, ln2_g + l * DD, ln2_b + l * DD, h32, h16);
  }
  gemm_kernel<0><<<dim3(4, 128), 256, 0, stream>>>(
      h16, wfin, nullptr, (float*)d_out, MTOK, DD, DD);
}
